// VLunchboxMHSA_45870250721699
// MI455X (gfx1250) — matrix operand order repaired and checked by reading, not compiled, not run on hardware
//
#include <hip/hip_runtime.h>

// MI455X (gfx1250): linear-attention reassociation
//   out = ( q @ [SCALE * K^T V] ) @ proj,  q,k from x@Wqkv
// ~11 GFLOP, ~100MB HBM traffic -> bandwidth bound at 23.3 TB/s.
//  - bf16 intermediates, v_wmma_f32_16x16x32_bf16 for all matmuls
//  - B operands stored column-major: A and B fragments are ds_load_b128
//  - double-buffered LDS filled with GLOBAL_LOAD_ASYNC_TO_LDS_B128
//    (ASYNCcnt/s_wait_asynccnt), overlapping next-tile DMA with WMMA
//  - global_prefetch_b8 two tiles ahead

typedef __bf16 bf16_t;
typedef bf16_t bf16x4  __attribute__((ext_vector_type(4)));
typedef bf16_t bf16x8  __attribute__((ext_vector_type(8)));
typedef bf16_t bf16x16 __attribute__((ext_vector_type(16)));
typedef float  f32x8   __attribute__((ext_vector_type(8)));

#define Bb 4
#define Nn 2048
#define Cc 512
#define Hh 8
#define Dd 64
#define Ll 64

// ---------------- async global -> LDS (CDNA5, ASYNCcnt) ---------------------

__device__ __forceinline__ unsigned lds_off32(const void* p) {
  // flat shared pointer: addr[31:0] is the workgroup-relative LDS address
  return (unsigned)(unsigned long long)p;
}

__device__ __forceinline__ void async_b128(unsigned lds, const void* gaddr) {
  asm volatile("global_load_async_to_lds_b128 %0, %1, off"
               :: "v"(lds), "v"(gaddr) : "memory");
}

__device__ __forceinline__ void wait_async0() {
  asm volatile("s_wait_asynccnt 0" ::: "memory");
}

// ---------------- WMMA fragment helpers (wave32, ISA 7.12.2 layouts) --------

// A 16x32 bf16 (row-major source): lane r / r+16 hold row M=r.
// elem e<8: K=hi*8+e ; elem e>=8: K=16+hi*8+(e-8)  -> two 16B loads.
__device__ __forceinline__ bf16x16 frag_a_row(const bf16_t* p, int ld) {
  const int lane = threadIdx.x & 31;
  const int r = lane & 15, hi = (lane >> 4) & 1;
  const bf16_t* q = p + r * ld + hi * 8;
  bf16x8 lo = *(const bf16x8*)(q);
  bf16x8 hh = *(const bf16x8*)(q + 16);
  return __builtin_shufflevector(lo, hh, 0, 1, 2, 3, 4, 5, 6, 7,
                                 8, 9, 10, 11, 12, 13, 14, 15);
}

// B 32x16 bf16 from COLUMN-major storage Bt[n][k] (stride ld):
// lane c holds col N=c, elem e<8: K=hi*8+e ; elem e>=8: K=16+hi*8+(e-8).
__device__ __forceinline__ bf16x16 frag_b_cm(const bf16_t* p, int ld) {
  const int lane = threadIdx.x & 31;
  const int c = lane & 15, hi = (lane >> 4) & 1;
  const bf16_t* q = p + c * ld + hi * 8;
  bf16x8 lo = *(const bf16x8*)(q);
  bf16x8 hh = *(const bf16x8*)(q + 16);
  return __builtin_shufflevector(lo, hh, 0, 1, 2, 3, 4, 5, 6, 7,
                                 8, 9, 10, 11, 12, 13, 14, 15);
}

// A layout but logical A[r][k] = p[k*ld + dcol + r] (transposed gather, used
// only for K^T in the tiny KV kernel).
__device__ __forceinline__ bf16x16 frag_a_transposed(const bf16_t* p, int ld, int dcol) {
  const int lane = threadIdx.x & 31;
  const int r = lane & 15, hi = (lane >> 4) & 1;
  bf16x16 a;
#pragma unroll
  for (int e = 0; e < 8; ++e) a[e] = p[(hi * 8 + e) * ld + dcol + r];
#pragma unroll
  for (int e = 0; e < 8; ++e) a[8 + e] = p[(16 + hi * 8 + e) * ld + dcol + r];
  return a;
}

__device__ __forceinline__ f32x8 wmma_bf16(bf16x16 a, bf16x16 b, f32x8 c) {
  return __builtin_amdgcn_wmma_f32_16x16x32_bf16(false, a, false, b,
                                                 (short)0, c, false, false);
}

// C/D 16x16 f32: VGPR i -> row hi*8+i, col = lane&15.
__device__ __forceinline__ void store_tile_bf16(bf16_t* d, int ldc, f32x8 c, float s) {
  const int lane = threadIdx.x & 31;
  const int col = lane & 15, hi = (lane >> 4) & 1;
#pragma unroll
  for (int i = 0; i < 8; ++i) d[(hi * 8 + i) * ldc + col] = (bf16_t)(c[i] * s);
}
__device__ __forceinline__ void store_tile_f32(float* d, int ldc, f32x8 c, float s) {
  const int lane = threadIdx.x & 31;
  const int col = lane & 15, hi = (lane >> 4) & 1;
#pragma unroll
  for (int i = 0; i < 8; ++i) d[(hi * 8 + i) * ldc + col] = c[i] * s;
}

// ---------------- conversions -----------------------------------------------

__global__ __launch_bounds__(256) void cvt_f32_to_bf16(const float* __restrict__ in,
                                                       bf16_t* __restrict__ out, int n) {
  int i = (blockIdx.x * 256 + threadIdx.x) * 4;
  if (i >= n) return;
  float4 f = *(const float4*)(in + i);
  bf16x4 o;
  o[0] = (bf16_t)f.x; o[1] = (bf16_t)f.y; o[2] = (bf16_t)f.z; o[3] = (bf16_t)f.w;
  *(bf16x4*)(out + i) = o;
}

// out[c][r] = (bf16) in[r][c]; in is R x C f32, out is C x R bf16. 32x32 tiles.
__global__ __launch_bounds__(256) void cvt_transpose_bf16(const float* __restrict__ in,
                                                          bf16_t* __restrict__ out,
                                                          int R, int C) {
  __shared__ bf16_t t[32][33];
  const int bx = blockIdx.x * 32;  // col base in `in`
  const int by = blockIdx.y * 32;  // row base in `in`
  const int tx = threadIdx.x & 31, ty = threadIdx.x >> 5;
#pragma unroll
  for (int i = 0; i < 32; i += 8)
    t[ty + i][tx] = (bf16_t)in[(long long)(by + ty + i) * C + (bx + tx)];
  __syncthreads();
#pragma unroll
  for (int i = 0; i < 32; i += 8)
    out[(long long)(bx + ty + i) * R + (by + tx)] = t[tx][ty + i];
}

// ---------------- generic batched bf16 WMMA GEMM (B column-major) -----------
// C[M,N] = scale * A[M,K] @ B[K,N], A row-major, B given as Bt[N][K].
// 128x64x64 block tiles, 256 thr = 8 waves, wave owns 32x32 (2x2 WMMA tiles),
// 8 wmma per k-step; double-buffered LDS filled by async-to-LDS DMA.

template <bool OUT_BF16>
__global__ __launch_bounds__(256) void gemm_bf16_wmma(
    const bf16_t* __restrict__ A, const bf16_t* __restrict__ Bt,
    void* __restrict__ Cv, int M, int N, int K, int lda, int ldbt, int ldc,
    int zdiv, long long sA1, long long sA2, long long sB1, long long sB2,
    long long sC1, long long sC2, float scale) {
  constexpr int BM = 128, BN = 64, BK = 64;
  constexpr int LDS_A = BK + 8;  // 72 halfwords: rows 144B (16B aligned)
  constexpr int LDS_B = BK + 8;
  __shared__ bf16_t As[2][BM * LDS_A];  // 2 x 18KB
  __shared__ bf16_t Bs[2][BN * LDS_B];  // 2 x  9KB

  const long long zb = blockIdx.z / zdiv;
  const long long zs = blockIdx.z % zdiv;
  A  += zb * sA1 + zs * sA2;
  Bt += zb * sB1 + zs * sB2;
  const long long coff = zb * sC1 + zs * sC2;

  const int bm = blockIdx.x * BM;
  const int bn = blockIdx.y * BN;
  const int tid = threadIdx.x;
  const int wave = tid >> 5;
  const int wm = wave >> 1;  // 0..3
  const int wn = wave & 1;   // 0..1

  f32x8 acc[2][2] = {};

  // issue 6 async-to-LDS b128 ops per thread (A: 128x64, Bt: 64x64)
  auto stage_async = [&](bf16_t* Asb, bf16_t* Bsb, int k0) {
#pragma unroll
    for (int it = 0; it < 4; ++it) {
      int linear = (it * 256 + tid) * 8;
      int r = linear >> 6, c = linear & 63;
      async_b128(lds_off32(&Asb[r * LDS_A + c]),
                 A + (long long)(bm + r) * lda + (k0 + c));
    }
#pragma unroll
    for (int it = 0; it < 2; ++it) {
      int linear = (it * 256 + tid) * 8;
      int r = linear >> 6, c = linear & 63;
      async_b128(lds_off32(&Bsb[r * LDS_B + c]),
                 Bt + (long long)(bn + r) * ldbt + (k0 + c));
    }
  };

  auto compute = [&](const bf16_t* Asb, const bf16_t* Bsb) {
#pragma unroll
    for (int kk = 0; kk < BK; kk += 32) {
      bf16x16 a0 = frag_a_row(&Asb[(wm * 32 + 0) * LDS_A + kk], LDS_A);
      bf16x16 a1 = frag_a_row(&Asb[(wm * 32 + 16) * LDS_A + kk], LDS_A);
      bf16x16 b0 = frag_b_cm(&Bsb[(wn * 32 + 0) * LDS_B + kk], LDS_B);
      bf16x16 b1 = frag_b_cm(&Bsb[(wn * 32 + 16) * LDS_B + kk], LDS_B);
      acc[0][0] = wmma_bf16(a0, b0, acc[0][0]);
      acc[0][1] = wmma_bf16(a0, b1, acc[0][1]);
      acc[1][0] = wmma_bf16(a1, b0, acc[1][0]);
      acc[1][1] = wmma_bf16(a1, b1, acc[1][1]);
    }
  };

  const int KT = K >> 6;
  stage_async(As[0], Bs[0], 0);
  for (int kt = 0; kt < KT; ++kt) {
    wait_async0();     // my stage(cur) writes have landed in LDS
    __syncthreads();   // everyone's landed; everyone done reading buf(next)
    const int cur = kt & 1;
    if (kt + 1 < KT)   // DMA next tile while we do WMMAs on this one
      stage_async(As[cur ^ 1], Bs[cur ^ 1], (kt + 1) * BK);
    if (kt + 2 < KT) {  // gfx1250 global_prefetch_b8, one line per tile row
      const int pk = (kt + 2) * BK;
      if (tid < 128)
        __builtin_prefetch((const void*)(A + (long long)(bm + tid) * lda + pk), 0, 1);
      else if (tid < 192)
        __builtin_prefetch((const void*)(Bt + (long long)(bn + tid - 128) * ldbt + pk), 0, 1);
    }
    compute(As[cur], Bs[cur]);
  }

#pragma unroll
  for (int i = 0; i < 2; ++i)
#pragma unroll
    for (int j = 0; j < 2; ++j) {
      const int row = bm + wm * 32 + i * 16;
      const int col = bn + wn * 32 + j * 16;
      if (OUT_BF16) {
        bf16_t* C = (bf16_t*)Cv + coff;
        store_tile_bf16(C + (long long)row * ldc + col, ldc, acc[i][j], scale);
      } else {
        float* C = (float*)Cv + coff;
        store_tile_f32(C + (long long)row * ldc + col, ldc, acc[i][j], scale);
      }
    }
}

// ---------------- KV kernel: KVt[b,h][l][d] = SCALE * (K^T V)^T -------------
// One block per (b,h); k tile (32m x 64d) + vT tile (64l x 32m) double-
// buffered in LDS via async DMA. A = K^T via LDS gather; B = vT col-major.
// Stores KV *transposed* so stage-3 B is column-major.

__global__ __launch_bounds__(256) void kv_wmma(const bf16_t* __restrict__ hbuf,
                                               const bf16_t* __restrict__ vt,
                                               bf16_t* __restrict__ kvt, float scale) {
  constexpr int LK = 72;  // ks row stride ([m=32][d=64])
  constexpr int LV = 40;  // vs row stride ([l=64][m=32])
  __shared__ bf16_t ks[2][32 * LK];
  __shared__ bf16_t vs[2][64 * LV];
  const int b = blockIdx.x >> 3;
  const int hh = blockIdx.x & 7;
  const bf16_t* kbase = hbuf + (long long)b * Nn * (2 * Hh * Dd) + hh * (2 * Dd) + Dd;
  const bf16_t* vbase = vt + (long long)hh * Ll * Nn;  // vT[l][m], ld = Nn

  const int tid = threadIdx.x;
  const int wave = tid >> 5;
  const int dt = wave >> 1;        // 16-row (d) tile, 0..3
  const int lt0 = (wave & 1) * 2;  // first of two 16-col (l) tiles
  f32x8 acc[2] = {};

  const int rk = tid >> 3, ck = (tid & 7) * 8;  // ks: 32 x 64
  const int rv = tid >> 2, cv = (tid & 3) * 8;  // vs: 64 x 32

  auto stage_async = [&](int buf, int m0) {
    async_b128(lds_off32(&ks[buf][rk * LK + ck]),
               kbase + (long long)(m0 + rk) * (2 * Hh * Dd) + ck);
    async_b128(lds_off32(&vs[buf][rv * LV + cv]),
               vbase + (long long)rv * Nn + (m0 + cv));
  };

  constexpr int MT = Nn / 32;
  stage_async(0, 0);
  for (int mt = 0; mt < MT; ++mt) {
    wait_async0();
    __syncthreads();
    const int cur = mt & 1;
    if (mt + 1 < MT) stage_async(cur ^ 1, (mt + 1) * 32);
    bf16x16 a  = frag_a_transposed(ks[cur], LK, dt * 16);
    bf16x16 b0 = frag_b_cm(&vs[cur][(lt0 + 0) * 16 * LV], LV);
    bf16x16 b1 = frag_b_cm(&vs[cur][(lt0 + 1) * 16 * LV], LV);
    acc[0] = wmma_bf16(a, b0, acc[0]);
    acc[1] = wmma_bf16(a, b1, acc[1]);
  }

  bf16_t* dst = kvt + (long long)blockIdx.x * (Dd * Ll);
  const int lane = tid & 31, col = lane & 15, hi = (lane >> 4) & 1;
#pragma unroll
  for (int j = 0; j < 2; ++j)
#pragma unroll
    for (int i = 0; i < 8; ++i)  // KVt[l][d] = KV[d][l]
      dst[((lt0 + j) * 16 + col) * Dd + (dt * 16 + hi * 8 + i)] =
          (bf16_t)(acc[j][i] * scale);
}

// ---------------- host launcher ---------------------------------------------

extern "C" void kernel_launch(void* const* d_in, const int* in_sizes, int n_in,
                              void* d_out, int out_size, void* d_ws, size_t ws_size,
                              hipStream_t stream) {
  (void)in_sizes; (void)n_in; (void)out_size; (void)ws_size;
  const float* x    = (const float*)d_in[0];   // (B,N,C)
  const float* v    = (const float*)d_in[1];   // (N,H,L) == (2048 x 512) flat
  const float* qkv  = (const float*)d_in[2];   // (C, 2*H*D) == (512 x 1024)
  const float* proj = (const float*)d_in[3];   // (H*L, D)  == (512 x 64)
  float* out = (float*)d_out;                  // (B,N,D)

  const long long nX  = (long long)Bb * Nn * Cc;           // 4,194,304
  const long long nV  = (long long)Nn * Hh * Ll;           // 1,048,576
  const long long nW  = (long long)Cc * 2 * Hh * Dd;       //   524,288
  const long long nP  = (long long)Hh * Ll * Dd;           //    32,768
  const long long nH  = (long long)Bb * Nn * 2 * Hh * Dd;  // 8,388,608
  const long long nKV = (long long)Bb * Hh * Dd * Ll;      //   131,072

  bf16_t* xb  = (bf16_t*)d_ws;   // x, row-major (8192 x 512)
  bf16_t* wt  = xb + nX;         // qkv^T      (1024 x 512)
  bf16_t* vt  = wt + nW;         // v^T        (512 x 2048) == vT[h][l][m]
  bf16_t* pt  = vt + nV;         // proj^T     (64 x 512)
  bf16_t* hb  = pt + nP;         // h = x@Wqkv (8192 x 1024)
  bf16_t* kvt = hb + nH;         // KV^T per (b,h), 64x64
  bf16_t* rb  = kvt + nKV;       // res        (8192 x 512)

  const float SCALE = 0.125f;  // 64^{-1/2}

  cvt_f32_to_bf16<<<dim3((unsigned)(nX / 1024)), 256, 0, stream>>>(x, xb, (int)nX);
  cvt_transpose_bf16<<<dim3(1024 / 32, 512 / 32), 256, 0, stream>>>(qkv, wt, 512, 1024);
  cvt_transpose_bf16<<<dim3(512 / 32, 2048 / 32), 256, 0, stream>>>(v, vt, 2048, 512);
  cvt_transpose_bf16<<<dim3(64 / 32, 512 / 32), 256, 0, stream>>>(proj, pt, 512, 64);

  // Stage 1: hb[8192x1024] = xb[8192x512] @ W[512x1024]   (B = wt col-major)
  gemm_bf16_wmma<true><<<dim3(8192 / 128, 1024 / 64, 1), 256, 0, stream>>>(
      xb, wt, hb, 8192, 1024, 512, 512, 512, 1024,
      1, 0, 0, 0, 0, 0, 0, 1.0f);

  // Stage 2: kvt[b,h] = SCALE * (K^T V)^T   (32 blocks)
  kv_wmma<<<dim3(32), 256, 0, stream>>>(hb, vt, kvt, SCALE);

  // Stage 3: rb[b,n, h*64+l] = q[b,n,h,:] @ KV[b,h]   (B = kvt col-major)
  gemm_bf16_wmma<true><<<dim3(2048 / 128, 1, 32), 256, 0, stream>>>(
      hb, kvt, rb, 2048, 64, 64, 1024, 64, 512,
      8, (long long)2048 * 1024, 128,
      (long long)8 * 4096, 4096,
      (long long)2048 * 512, 64, 1.0f);

  // Stage 4: out[8192x64] = rb[8192x512] @ proj[512x64]  (B = pt col-major)
  gemm_bf16_wmma<false><<<dim3(8192 / 128, 1, 1), 256, 0, stream>>>(
      rb, pt, out, 8192, 64, 512, 512, 512, 64,
      1, 0, 0, 0, 0, 0, 0, 1.0f);
}
